// AttentionalSampling_15539191677460
// MI455X (gfx1250) — matrix operand order repaired and checked by reading, not compiled, not run on hardware
//
#include <hip/hip_runtime.h>

typedef __attribute__((ext_vector_type(16))) __bf16   v16bf;
typedef __attribute__((ext_vector_type(8)))  float    v8f;
typedef __attribute__((ext_vector_type(8)))  unsigned u32x8;
typedef __attribute__((ext_vector_type(4)))  unsigned u32x4;
typedef __attribute__((ext_vector_type(8)))  int      i32x8;
typedef __attribute__((ext_vector_type(4)))  int      i32x4;

#define WMMA_BF16(A,B,C) \
  __builtin_amdgcn_wmma_f32_16x16x32_bf16(false,(A),false,(B),(short)0,(C),false,false)

#if defined(__HIP_DEVICE_COMPILE__) && defined(__has_builtin)
#if __has_builtin(__builtin_amdgcn_tensor_load_to_lds)
#define HAVE_TDM 1
#endif
#endif

__device__ __forceinline__ unsigned short f2bf(float f) {
  unsigned u = __builtin_bit_cast(unsigned, f);
  unsigned r = (u + 0x7FFFu + ((u >> 16) & 1u)) >> 16;   // round-to-nearest-even
  return (unsigned short)r;
}
__device__ __forceinline__ unsigned pack2(float lo, float hi) {
  return (unsigned)f2bf(lo) | ((unsigned)f2bf(hi) << 16);
}
__device__ __forceinline__ v8f zero8() {
  v8f z; for (int i = 0; i < 8; ++i) z[i] = 0.0f; return z;
}
// 16-element bf16 WMMA fragment as two 16B chunks (ds/global b128 loads)
__device__ __forceinline__ v16bf ld_frag(const unsigned short* p0,
                                         const unsigned short* p1) {
  uint4 lo = *(const uint4*)p0;
  uint4 hi = *(const uint4*)p1;
  u32x8 u;
  u[0] = lo.x; u[1] = lo.y; u[2] = lo.z; u[3] = lo.w;
  u[4] = hi.x; u[5] = hi.y; u[6] = hi.z; u[7] = hi.w;
  return __builtin_bit_cast(v16bf, u);
}

#ifdef HAVE_TDM
// Issue a TDM 2D tile load: 64 rows x 96 bf16, row stride 768 elements.
__device__ __forceinline__ void tdm_load_k64x96(const unsigned short* gsrc,
                                                unsigned lds_addr) {
  unsigned long long ga = (unsigned long long)(uintptr_t)gsrc;
  u32x4 g0;
  g0[0] = 1u;                                   // count=1 (valid user D#)
  g0[1] = lds_addr;                             // lds_addr (bytes)
  g0[2] = (unsigned)ga;                         // global_addr[31:0]
  g0[3] = (unsigned)(ga >> 32) | (2u << 30);    // global_addr hi | type=2
  i32x8 g1;
  g1[0] = (1 << 16);                            // data_size=1 (2B), wg_mask=0
  g1[1] = (int)((768u & 0xFFFFu) << 16);        // tensor_dim0 lo16
  g1[2] = (int)((768u >> 16) | ((16384u & 0xFFFFu) << 16)); // dim0 hi | dim1 lo
  g1[3] = (int)((16384u >> 16) | (96u << 16));  // dim1 hi | tile_dim0=96
  g1[4] = 64;                                   // tile_dim1=64, tile_dim2=0
  g1[5] = 768;                                  // tensor_dim0_stride lo32
  g1[6] = 0;
  g1[7] = 0;
  i32x4 zg = {0, 0, 0, 0};                      // groups 2/3 unused (2D)
#if defined(__clang_major__) && __clang_major__ >= 23
  i32x8 zg8 = {0, 0, 0, 0, 0, 0, 0, 0};
  __builtin_amdgcn_tensor_load_to_lds(g0, g1, zg, zg, zg8, 0);
#else
  __builtin_amdgcn_tensor_load_to_lds(g0, g1, zg, zg, 0);
#endif
}
#endif

// ---------------------------------------------------------------------------
// Tiled GEMM: C[M,N] = (A (+A2?))[M,K] * B[K,N]; fp32 in, bf16 WMMA, fp32 acc.
// Block 256 (8 waves); C macro-tile 128x64; each wave 32x32 (4 WMMA tiles).
// ---------------------------------------------------------------------------
template <int ADD_A2, int OUT_BF16>
__global__ __launch_bounds__(256) void gemm_bf16_kernel(
    const float* __restrict__ A, const float* __restrict__ A2,
    const float* __restrict__ B, float* __restrict__ Cf,
    unsigned short* __restrict__ Cb, int M, int N, int K)
{
  __shared__ __attribute__((aligned(16))) unsigned short As[128][32]; // rows x k
  __shared__ __attribute__((aligned(16))) unsigned short Bt[64][32];  // cols x k

  const int tid  = threadIdx.x;
  const int lane = tid & 31;
  const int wave = tid >> 5;
  const int bm = blockIdx.y * 128, bn = blockIdx.x * 64;
  const int wm = (wave & 3) * 32, wn = (wave >> 2) * 32;
  const int l15 = lane & 15;
  const int hi  = lane >> 4;
  const int aB1 = hi * 8;        // unused (A fragments load at bB)
  const int bB  = hi * 16;       // A/B fragment k base (contiguous k = 16h + s)

  const int am  = tid >> 2;            // A: row 0..63 (plus +64 chunk)
  const int ak8 = (tid & 3) << 3;      // A: k chunk of 8
  const int bkp = tid >> 4;            // B: k-pair 0..15 (k = 2*bkp)
  const int bn4 = (tid & 15) << 2;     // B: 4 columns

  v8f c00 = zero8(), c01 = zero8(), c10 = zero8(), c11 = zero8();

  for (int kk = 0; kk < K; kk += 32) {
    // stage A tile 128x32 (two row-chunks per thread)
    for (int cA = 0; cA < 2; ++cA) {
      int m = am + cA * 64;
      const float* ap = A + (size_t)(bm + m) * K + kk + ak8;
      float4 x = *(const float4*)ap;
      float4 y = *(const float4*)(ap + 4);
      if (ADD_A2) {
        const float* a2p = A2 + (size_t)(bm + m) * K + kk + ak8;
        float4 x2 = *(const float4*)a2p;
        float4 y2 = *(const float4*)(a2p + 4);
        x.x += x2.x; x.y += x2.y; x.z += x2.z; x.w += x2.w;
        y.x += y2.x; y.y += y2.y; y.z += y2.z; y.w += y2.w;
      }
      uint4 pk;
      pk.x = pack2(x.x, x.y); pk.y = pack2(x.z, x.w);
      pk.z = pack2(y.x, y.y); pk.w = pack2(y.z, y.w);
      *(uint4*)&As[m][ak8] = pk;
    }
    // stage B tile 32x64 transposed: pack two adjacent k rows -> b32 stores
    {
      const float* bp = B + (size_t)(kk + 2 * bkp) * N + bn + bn4;
      float4 r0 = *(const float4*)bp;
      float4 r1 = *(const float4*)(bp + N);
      *(unsigned*)&Bt[bn4 + 0][2 * bkp] = pack2(r0.x, r1.x);
      *(unsigned*)&Bt[bn4 + 1][2 * bkp] = pack2(r0.y, r1.y);
      *(unsigned*)&Bt[bn4 + 2][2 * bkp] = pack2(r0.z, r1.z);
      *(unsigned*)&Bt[bn4 + 3][2 * bkp] = pack2(r0.w, r1.w);
    }
    if (kk + 32 < K) {   // CDNA5 global_prefetch_b8 for next tiles
      __builtin_prefetch(&A[(size_t)(bm + am) * K + kk + 32], 0, 0);
      __builtin_prefetch(&B[(size_t)(kk + 32 + bkp) * N + bn], 0, 0);
    }
    __syncthreads();

    v16bf a0 = ld_frag(&As[wm + l15][bB],      &As[wm + l15][bB + 8]);
    v16bf a1 = ld_frag(&As[wm + 16 + l15][bB], &As[wm + 16 + l15][bB + 8]);
    v16bf b0 = ld_frag(&Bt[wn + l15][bB],       &Bt[wn + l15][bB + 8]);
    v16bf b1 = ld_frag(&Bt[wn + 16 + l15][bB],  &Bt[wn + 16 + l15][bB + 8]);
    c00 = WMMA_BF16(a0, b0, c00);
    c01 = WMMA_BF16(a0, b1, c01);
    c10 = WMMA_BF16(a1, b0, c10);
    c11 = WMMA_BF16(a1, b1, c11);
    __syncthreads();
  }

  for (int i = 0; i < 8; ++i) {
    int row0 = bm + wm + i + hi * 8;
    int row1 = row0 + 16;
    int col0 = bn + wn + l15;
    if (OUT_BF16) {
      Cb[(size_t)row0 * N + col0]      = f2bf(c00[i]);
      Cb[(size_t)row0 * N + col0 + 16] = f2bf(c01[i]);
      Cb[(size_t)row1 * N + col0]      = f2bf(c10[i]);
      Cb[(size_t)row1 * N + col0 + 16] = f2bf(c11[i]);
    } else {
      Cf[(size_t)row0 * N + col0]      = c00[i];
      Cf[(size_t)row0 * N + col0 + 16] = c01[i];
      Cf[(size_t)row1 * N + col0]      = c10[i];
      Cf[(size_t)row1 * N + col0 + 16] = c11[i];
    }
  }
}

// ---------------------------------------------------------------------------
// LayerNorm over 768-dim rows (eps 1e-6, weight only), fp32 in -> bf16 out
// ---------------------------------------------------------------------------
__global__ __launch_bounds__(256) void ln_kernel(
    const float* __restrict__ X, const float* __restrict__ w,
    unsigned short* __restrict__ Y)
{
  const int row = blockIdx.x, tid = threadIdx.x;
  const float* x = X + (size_t)row * 768;
  float v0 = x[tid], v1 = x[tid + 256], v2 = x[tid + 512];
  __shared__ float sh1[256], sh2[256];
  sh1[tid] = v0 + v1 + v2;
  sh2[tid] = v0 * v0 + v1 * v1 + v2 * v2;
  __syncthreads();
  for (int off = 128; off > 0; off >>= 1) {
    if (tid < off) { sh1[tid] += sh1[tid + off]; sh2[tid] += sh2[tid + off]; }
    __syncthreads();
  }
  float mean = sh1[0] * (1.0f / 768.0f);
  float var  = sh2[0] * (1.0f / 768.0f) - mean * mean;
  float rs   = rsqrtf(var + 1e-6f);
  unsigned short* y = Y + (size_t)row * 768;
  y[tid]       = f2bf((v0 - mean) * rs * w[tid]);
  y[tid + 256] = f2bf((v1 - mean) * rs * w[tid + 256]);
  y[tid + 512] = f2bf((v2 - mean) * rs * w[tid + 512]);
}

// ---------------------------------------------------------------------------
// Flash attention per (t,h). K tiles double-buffered through the Tensor Data
// Mover (prefetch tile kt+1 while computing on kt, TENSORcnt-throttled).
// ---------------------------------------------------------------------------
__global__ __launch_bounds__(256) void attn_kernel(
    const unsigned short* __restrict__ Qbf, const unsigned short* __restrict__ Kbf,
    const unsigned short* __restrict__ Vbf, const float* __restrict__ tracks,
    const float* __restrict__ fpos, float* __restrict__ samp)
{
  __shared__ __attribute__((aligned(16))) unsigned short Ks[2][64][96];  // ping-pong
  __shared__ __attribute__((aligned(16))) unsigned short Vt[96][64];     // d x keys
  __shared__ __attribute__((aligned(16))) unsigned short Ps[8][16][64];  // per-wave P
  __shared__ float fps[64][2];
  __shared__ float trk[256][2];

  const int bx = blockIdx.x;
  const int t = bx >> 3, h = bx & 7;
  const int tid  = threadIdx.x;
  const int lane = tid & 31, wave = tid >> 5;
  const int l15 = lane & 15, hi = lane >> 4;
  const int aB1 = hi * 8, bB = hi * 16;
  const int mr = wave * 32;

  trk[tid][0] = tracks[((size_t)t * 256 + tid) * 2 + 0];
  trk[tid][1] = tracks[((size_t)t * 256 + tid) * 2 + 1];

  // Q fragments in registers (global b128 pairs)
  v16bf qf[2][3];
  for (int rt = 0; rt < 2; ++rt)
    for (int ks = 0; ks < 3; ++ks) {
      int m = mr + rt * 16 + l15;
      const unsigned short* qrow =
          Qbf + ((size_t)t * 256 + m) * 768 + h * 96 + ks * 32;
      qf[rt][ks] = ld_frag(qrow + bB, qrow + bB + 8);
    }

  float mu[2][8], ls[2][8];
  v8f o[2][6];
  for (int rt = 0; rt < 2; ++rt) {
    for (int i = 0; i < 8; ++i) { mu[rt][i] = -1e30f; ls[rt][i] = 0.0f; }
    for (int ct = 0; ct < 6; ++ct) o[rt][ct] = zero8();
  }

  const float sscale = 0.10206207262f;   // 1/sqrt(96)
  const unsigned short* kbase = Kbf + (size_t)t * 1024 * 768 + h * 96;

#ifdef HAVE_TDM
  const unsigned lds_ks0 = (unsigned)(unsigned long long)(uintptr_t)&Ks[0][0][0];
  if (wave == 0)           // prefetch first K tile
    tdm_load_k64x96(kbase, lds_ks0);
#endif

  for (int kt = 0; kt < 16; ++kt) {
    const int n0 = kt * 64;
    const int buf = kt & 1;

    // V tile: vectorized read, transposed scatter into LDS
    for (int i = 0; i < 3; ++i) {
      int c = tid + i * 256;
      int key = c / 12, d8 = (c % 12) * 8;
      uint4 v = *(const uint4*)(Vbf + ((size_t)t * 1024 + n0 + key) * 768 + h * 96 + d8);
      unsigned short e[8];
      *(uint4*)e = v;
      for (int j = 0; j < 8; ++j) Vt[d8 + j][key] = e[j];
    }
    if (tid < 64) {
      fps[tid][0] = fpos[(size_t)(n0 + tid) * 2 + 0];
      fps[tid][1] = fpos[(size_t)(n0 + tid) * 2 + 1];
    }

#ifdef HAVE_TDM
    if (wave == 0) {
      if (kt + 1 < 16) {   // issue next tile, then retire current (in-order)
        tdm_load_k64x96(kbase + (size_t)(n0 + 64) * 768,
                        lds_ks0 + ((kt + 1) & 1) * (64 * 96 * 2));
        __builtin_amdgcn_s_wait_tensorcnt(1);
      } else {
        __builtin_amdgcn_s_wait_tensorcnt(0);
      }
    }
#else
    for (int i = 0; i < 3; ++i) {        // fallback: manual K tile
      int c = tid + i * 256;
      int key = c / 12, d8 = (c % 12) * 8;
      *(uint4*)&Ks[buf][key][d8] =
          *(const uint4*)(kbase + (size_t)(n0 + key) * 768 + d8);
    }
#endif
    __syncthreads();

    for (int rt = 0; rt < 2; ++rt) {
      // ---- S = Q * K^T (WMMA-K = head dim, 3 steps) ----
      v8f s[4];
      for (int nt = 0; nt < 4; ++nt) {
        s[nt] = zero8();
        for (int ks = 0; ks < 3; ++ks) {
          const unsigned short* krow = &Ks[buf][nt * 16 + l15][ks * 32 + bB];
          v16bf kb = ld_frag(krow, krow + 8);
          s[nt] = WMMA_BF16(qf[rt][ks], kb, s[nt]);
        }
      }
      // ---- scale + Gaussian bias; online softmax ----
      float fx[4], fy[4];
      for (int nt = 0; nt < 4; ++nt) {
        fx[nt] = fps[nt * 16 + l15][0];
        fy[nt] = fps[nt * 16 + l15][1];
      }
      float st[4][8], pm[8];
      for (int i = 0; i < 8; ++i) {
        int m = mr + rt * 16 + i + hi * 8;
        float tx = trk[m][0], ty = trk[m][1];
        float mx = -1e30f;
        for (int nt = 0; nt < 4; ++nt) {
          float dx = tx - fx[nt], dy = ty - fy[nt];
          float v = s[nt][i] * sscale - 2.0f * (dx * dx + dy * dy); // 1/(2*sigma^2)
          st[nt][i] = v;
          mx = fmaxf(mx, v);
        }
        pm[i] = mx;
      }
      for (int i = 0; i < 8; ++i)
        for (int off = 1; off < 16; off <<= 1)
          pm[i] = fmaxf(pm[i], __shfl_xor(pm[i], off, 16));
      float al[8];
      for (int i = 0; i < 8; ++i) {
        float mn = fmaxf(mu[rt][i], pm[i]);
        al[i] = __expf(mu[rt][i] - mn);
        mu[rt][i] = mn;
      }
      float rsum[8];
      for (int i = 0; i < 8; ++i) {
        float acc = 0.0f;
        for (int nt = 0; nt < 4; ++nt) {
          float p = __expf(st[nt][i] - mu[rt][i]);
          acc += p;
          Ps[wave][i + hi * 8][nt * 16 + l15] = f2bf(p);
        }
        rsum[i] = acc;
      }
      for (int i = 0; i < 8; ++i)
        for (int off = 1; off < 16; off <<= 1)
          rsum[i] += __shfl_xor(rsum[i], off, 16);
      for (int i = 0; i < 8; ++i) ls[rt][i] = ls[rt][i] * al[i] + rsum[i];
      for (int ct = 0; ct < 6; ++ct)
        for (int i = 0; i < 8; ++i) o[rt][ct][i] *= al[i];

      asm volatile("" ::: "memory");  // wave-local P: DS ops in-order per wave

      // ---- O += P * V (WMMA-K = key index; V transposed in LDS) ----
      for (int ks2 = 0; ks2 < 2; ++ks2) {
        const unsigned short* prow = &Ps[wave][l15][ks2 * 32];
        v16bf pa = ld_frag(prow + bB, prow + bB + 8);
        for (int ct = 0; ct < 6; ++ct) {
          const unsigned short* vrow = &Vt[ct * 16 + l15][ks2 * 32 + bB];
          v16bf vb = ld_frag(vrow, vrow + 8);
          o[rt][ct] = WMMA_BF16(pa, vb, o[rt][ct]);
        }
      }
    }
    __syncthreads();
  }

  for (int rt = 0; rt < 2; ++rt) {
    float inv[8];
    for (int i = 0; i < 8; ++i) inv[i] = 1.0f / ls[rt][i];
    for (int ct = 0; ct < 6; ++ct)
      for (int i = 0; i < 8; ++i) {
        int m = mr + rt * 16 + i + hi * 8;
        samp[((size_t)t * 256 + m) * 768 + h * 96 + ct * 16 + l15] =
            o[rt][ct][i] * inv[i];
      }
  }
}

// ---------------------------------------------------------------------------
extern "C" void kernel_launch(void* const* d_in, const int* in_sizes, int n_in,
                              void* d_out, int out_size, void* d_ws, size_t ws_size,
                              hipStream_t stream) {
  (void)in_sizes; (void)n_in; (void)out_size; (void)ws_size;
  const float* features = (const float*)d_in[0];   // [16,1024,768]
  const float* tracks   = (const float*)d_in[1];   // [16,256,2]
  const float* tpe      = (const float*)d_in[2];   // [16,256,768]
  const float* fpe      = (const float*)d_in[3];   // [16,1024,768]
  const float* fposi    = (const float*)d_in[4];   // [1024,2]
  const float* Wq       = (const float*)d_in[5];
  const float* Wk       = (const float*)d_in[6];
  const float* Wv       = (const float*)d_in[7];
  const float* Wo       = (const float*)d_in[8];
  const float* qlnw     = (const float*)d_in[9];
  const float* klnw     = (const float*)d_in[10];

  char* p = (char*)d_ws;
  float* qpre = (float*)p;          p += (size_t)4096  * 768 * 4;
  float* kpre = (float*)p;          p += (size_t)16384 * 768 * 4;
  float* smp  = (float*)p;          p += (size_t)4096  * 768 * 4;
  unsigned short* qbf = (unsigned short*)p; p += (size_t)4096  * 768 * 2;
  unsigned short* kbf = (unsigned short*)p; p += (size_t)16384 * 768 * 2;
  unsigned short* vbf = (unsigned short*)p; p += (size_t)16384 * 768 * 2;

  dim3 blk(256);
  gemm_bf16_kernel<0,0><<<dim3(12, 32),  blk, 0, stream>>>(tpe, nullptr, Wq, qpre, nullptr, 4096, 768, 768);
  gemm_bf16_kernel<1,0><<<dim3(12, 128), blk, 0, stream>>>(features, fpe, Wk, kpre, nullptr, 16384, 768, 768);
  gemm_bf16_kernel<0,1><<<dim3(12, 128), blk, 0, stream>>>(features, nullptr, Wv, nullptr, vbf, 16384, 768, 768);
  ln_kernel<<<4096,  256, 0, stream>>>(qpre, qlnw, qbf);
  ln_kernel<<<16384, 256, 0, stream>>>(kpre, klnw, kbf);
  attn_kernel<<<128, 256, 0, stream>>>(qbf, kbf, vbf, tracks, fposi, smp);
  gemm_bf16_kernel<0,0><<<dim3(12, 32), blk, 0, stream>>>(smp, nullptr, Wo, (float*)d_out, nullptr, 4096, 768, 768);
}
